// GuidedCxtAtten_52862457479802
// MI455X (gfx1250) — compile-verified, matrix operand order checked by reading, not run on hardware
//
#include <hip/hip_runtime.h>
#include <hip/hip_bf16.h>

typedef __attribute__((ext_vector_type(16))) _Float16 v16h;
typedef __attribute__((ext_vector_type(8)))  float    v8f;

#define NB   4      // batch
#define CG   128    // f channels
#define CO   64     // alpha/out channels
#define HH   128
#define WW   128
#define HS   64     // h = H/rate
#define WS   64
#define LL   4096   // h*w
#define K9   576    // CO*9
#define VD   1024   // CO*16
#define QS   512    // query stripe width
#define NSTR 8      // LL/QS

// ---------------- WMMA fragment helpers (wave32, CDNA5 layouts) -------------
// 16x32 f16 tile, source stored [row][k] (row = M for A, N for B^T), k contiguous.
__device__ inline v16h frag_ld_kcontig(const _Float16* p, int stride) {
  int lane = threadIdx.x & 31;
  int r  = lane & 15;
  int kb = (lane & 16) ? 8 : 0;
  const _Float16* s = p + (size_t)r * stride + kb;
  v16h v;
#pragma unroll
  for (int e = 0; e < 8; ++e) { v[e] = s[e]; v[8 + e] = s[16 + e]; }
  return v;
}

__device__ inline v8f wmma_f16(v16h a, v16h b, v8f c) {
  return __builtin_amdgcn_wmma_f32_16x16x32_f16(false, a, false, b, (short)0, c,
                                                false, false);
}

// Async global -> LDS copy, 16 bytes per lane (CDNA5, tracked by ASYNCcnt).
__device__ inline void async_cp_b128(unsigned lds_off, unsigned long long gaddr) {
  asm volatile("global_load_async_to_lds_b128 %0, %1, off"
               :: "v"(lds_off), "v"(gaddr) : "memory");
}

// ---------------- K0: convert weights to f16, zero BN stats ------------------
__global__ void k_init(const float* __restrict__ gw, const float* __restrict__ Wc,
                       _Float16* gwh, _Float16* wch, float* bns) {
  int t = blockIdx.x * blockDim.x + threadIdx.x;
  if (t < CO * CG) gwh[t] = (_Float16)gw[t];
  if (t < CO * CO) wch[t] = (_Float16)Wc[t];
  if (t < 2 * CO)  bns[t] = 0.0f;
}

// ---------------- K1: fd = (gw @ f)[::2,::2] + gb  (WMMA 64x128xpix) ---------
__global__ void k_fd(const float* __restrict__ f, const _Float16* __restrict__ gwh,
                     const float* __restrict__ gb, float* __restrict__ fd) {
  __shared__ _Float16 lb[16 * CG];           // [pix][c]  (B^T, N-major)
  int b  = blockIdx.y;
  int p0 = blockIdx.x * 16;
  int t  = threadIdx.x;                      // 128 threads, 4 waves
  for (int i = t; i < 16 * CG; i += 128) {
    int pix = i >> 7, c = i & (CG - 1);
    int p = p0 + pix, y = p >> 6, x = p & 63;
    lb[i] = (_Float16)f[((size_t)(b * CG + c) * HH + 2 * y) * WW + 2 * x];
  }
  __syncthreads();
  int wv = t >> 5;
  int o0 = wv * 16;
  v8f acc = {};
#pragma unroll
  for (int kc = 0; kc < CG; kc += 32) {
    v16h a  = frag_ld_kcontig(gwh + (size_t)o0 * CG + kc, CG);
    v16h bb = frag_ld_kcontig(lb + kc, CG);
    acc = wmma_f16(a, bb, acc);
  }
  int lane = t & 31, n = lane & 15, mb = (lane & 16) ? 8 : 0;
  int p = p0 + n, y = p >> 6, x = p & 63;
#pragma unroll
  for (int r = 0; r < 8; ++r) {
    int o = o0 + mb + r;
    fd[((size_t)(b * CO + o) * HS + y) * WS + x] = acc[r] + gb[o];
  }
}

// ---------------- K2: 3x3 reflect patches + L2 normalize ---------------------
__global__ void k_patch(const float* __restrict__ fd, _Float16* __restrict__ wp,
                        _Float16* __restrict__ wn) {
  __shared__ float vals[K9];
  __shared__ float red[8];
  int bp = blockIdx.x, b = blockIdx.y;
  int y = bp >> 6, x = bp & 63;
  int t = threadIdx.x;                       // 256 threads
  float ss = 0.0f;
  for (int e = t; e < K9; e += 256) {
    int c = e / 9, dd = e % 9, dy = dd / 3, dx = dd % 3;
    int ry = y + dy - 1; ry = ry < 0 ? -ry : (ry >= HS ? 2 * HS - 2 - ry : ry);
    int rx = x + dx - 1; rx = rx < 0 ? -rx : (rx >= WS ? 2 * WS - 2 - rx : rx);
    float v = fd[((size_t)(b * CO + c) * HS + ry) * WS + rx];
    vals[e] = v;
    ss += v * v;
  }
#pragma unroll
  for (int m = 16; m >= 1; m >>= 1) ss += __shfl_xor(ss, m, 32);
  if ((t & 31) == 0) red[t >> 5] = ss;
  __syncthreads();
  float tot = 0.0f;
#pragma unroll
  for (int i = 0; i < 8; ++i) tot += red[i];
  float inv = 1.0f / fmaxf(sqrtf(tot), 1e-4f);
  for (int e = t; e < K9; e += 256) {
    float v = vals[e];
    size_t idx = ((size_t)(b * LL + bp)) * K9 + e;
    wp[idx] = (_Float16)v;
    wn[idx] = (_Float16)(v * inv);
  }
}

// ---------------- K3: value patches, transposed ApT[b][v][p] (v-major) -------
__global__ void k_ap(const float* __restrict__ alpha, _Float16* __restrict__ apT) {
  const long total = (long)NB * VD * LL;
  for (long i = (long)blockIdx.x * blockDim.x + threadIdx.x; i < total;
       i += (long)gridDim.x * blockDim.x) {
    int p = (int)(i & (LL - 1));
    long r = i >> 12;
    int col = (int)(r & (VD - 1));
    int b = (int)(r >> 10);
    int c = col >> 4, ky = (col >> 2) & 3, kx = col & 3;
    int y = p >> 6, x = p & 63;
    int ry = 2 * y + ky - 1; ry = ry < 0 ? -ry : (ry >= HH ? 2 * HH - 2 - ry : ry);
    int rx = 2 * x + kx - 1; rx = rx < 0 ? -rx : (rx >= WW ? 2 * WW - 2 - rx : rx);
    apT[i] = (_Float16)(0.25f * alpha[((size_t)(b * CO + c) * HH + ry) * WW + rx]);
  }
}

// ---------------- K4: scores stripe  S[p, q'] = Wn[p]·Wp[q0+q'] --------------
// 8 waves x 2 p-tiles per wave; query tile staged in LDS.
__global__ void k_qk(const _Float16* __restrict__ wn, const _Float16* __restrict__ wp,
                     float* __restrict__ sbuf, int q0) {
  __shared__ _Float16 qt[16 * K9];           // [q][k]
  int b   = blockIdx.z;
  int qt0 = q0 + blockIdx.y * 16;
  int t   = threadIdx.x;                     // 256 threads
  const _Float16* wpb = wp + (size_t)b * LL * K9;
  for (int i = t; i < 16 * K9; i += 256) {
    int qq = i / K9, kk = i % K9;
    qt[i] = wpb[(size_t)(qt0 + qq) * K9 + kk];
  }
  __syncthreads();
  int wv = t >> 5;
  int pt = (blockIdx.x * 8 + wv) * 2;        // first of 2 p-tiles
  const _Float16* wnb = wn + (size_t)b * LL * K9;
  v8f acc[2] = {};
#pragma unroll
  for (int kc = 0; kc < K9; kc += 32) {
    v16h bb = frag_ld_kcontig(qt + kc, K9);
    v16h a0 = frag_ld_kcontig(wnb + (size_t)(pt + 0) * 16 * K9 + kc, K9);
    v16h a1 = frag_ld_kcontig(wnb + (size_t)(pt + 1) * 16 * K9 + kc, K9);
    acc[0] = wmma_f16(a0, bb, acc[0]);
    acc[1] = wmma_f16(a1, bb, acc[1]);
  }
  int lane = t & 31, n = lane & 15, mb = (lane & 16) ? 8 : 0;
  int qg = qt0 + n;
  int qs = blockIdx.y * 16 + n;
#pragma unroll
  for (int tt = 0; tt < 2; ++tt) {
    int ptc = pt + tt;
    float* sp = sbuf + ((size_t)b * LL + (size_t)ptc * 16) * QS + qs;
#pragma unroll
    for (int r = 0; r < 8; ++r) {
      int pg = ptc * 16 + mb + r;
      float v = acc[tt][r];
      if (pg == qg) v -= 10000.0f;
      sp[(size_t)(mb + r) * QS] = v;
    }
  }
}

// ---------------- K5: column softmax (unnormalized exp) + 1/sum -------------
// Block handles 16 columns; coalesced writes via LDS transpose staging.
__global__ void k_sm(const float* __restrict__ sbuf, _Float16* __restrict__ at,
                     float* __restrict__ invb) {
  __shared__ float red[16][16];
  __shared__ float colmax[16];
  __shared__ __align__(16) _Float16 eT[16 * 256];
  int b = blockIdx.y;
  int q0l = blockIdx.x * 16;
  int t = threadIdx.x;                       // 256 threads
  int ql = t & 15, pr = t >> 4;
  const float* sb = sbuf + (size_t)b * LL * QS + q0l + ql;
  // pass 1: column max
  float mx = -1e30f;
  for (int p = pr; p < LL; p += 16) mx = fmaxf(mx, sb[(size_t)p * QS]);
  red[pr][ql] = mx;
  __syncthreads();
  for (int s = 8; s >= 1; s >>= 1) {
    if (pr < s) red[pr][ql] = fmaxf(red[pr][ql], red[pr + s][ql]);
    __syncthreads();
  }
  if (pr == 0) colmax[ql] = red[0][ql];
  __syncthreads();
  float bm = colmax[ql];
  // pass 2: exp, stage transposed in LDS, coalesced writes; accumulate sum
  float s = 0.0f;
  _Float16* dst = at + ((size_t)b * QS + q0l) * LL;
  for (int pc = 0; pc < LL; pc += 256) {
    float ev[16];
#pragma unroll
    for (int j = 0; j < 16; ++j) {
      float v = __expf(sb[(size_t)(pc + pr + 16 * j) * QS] - bm);
      s += v;
      ev[j] = v;
    }
#pragma unroll
    for (int j = 0; j < 16; ++j) eT[ql * 256 + pr + 16 * j] = (_Float16)ev[j];
    __syncthreads();
    for (int idx = t; idx < 16 * 256; idx += 256) {
      int qr = idx >> 8, pl = idx & 255;
      dst[(size_t)qr * LL + pc + pl] = eT[idx];
    }
    __syncthreads();
  }
  red[pr][ql] = s;
  __syncthreads();
  for (int sd = 8; sd >= 1; sd >>= 1) {
    if (pr < sd) red[pr][ql] += red[pr + sd][ql];
    __syncthreads();
  }
  if (pr == 0) invb[(size_t)b * QS + q0l + ql] = 1.0f / red[0][ql];
}

// ---------------- K6: PV GEMM out[q,v] = inv[q] * sum_p e[q,p]*ApT[v,p] ------
// Async double-buffered LDS staging of the attention tile; 4 v-tiles per wave.
__global__ void k_pv(const _Float16* __restrict__ at, const _Float16* __restrict__ apT,
                     const float* __restrict__ invb, float* __restrict__ pv, int q0) {
  __shared__ __align__(16) _Float16 atile[2][16 * 256];
  int b = blockIdx.z, qt = blockIdx.y;
  int t = threadIdx.x;                       // 256 threads, 8 waves
  int wv = t >> 5;
  int v0 = (blockIdx.x * 8 + wv) * 64;       // 4 v-tiles of 16
  const _Float16* atb = at + ((size_t)b * QS + (size_t)qt * 16) * LL;
  const _Float16* apb = apT + (size_t)b * VD * LL;
  int qq = t >> 5;                           // 0..7 (row pairs qq, qq+8)
  int pl8 = (t & 31) * 8;                    // 8-half (16B) segment within chunk
  auto issue = [&](int buf, int pc) {
    unsigned d0 = (unsigned)(unsigned long long)&atile[buf][qq * 256 + pl8];
    unsigned long long s0 =
        (unsigned long long)(const void*)(atb + (size_t)qq * LL + pc + pl8);
    async_cp_b128(d0, s0);
    unsigned d1 = (unsigned)(unsigned long long)&atile[buf][(qq + 8) * 256 + pl8];
    unsigned long long s1 =
        (unsigned long long)(const void*)(atb + (size_t)(qq + 8) * LL + pc + pl8);
    async_cp_b128(d1, s1);
  };
  v8f acc[4] = {};
  issue(0, 0);
  for (int i = 0; i < LL / 256; ++i) {
    int pc = i * 256;
    if (i + 1 < LL / 256) {
      issue((i + 1) & 1, pc + 256);
      asm volatile("s_wait_asynccnt 0x2" ::: "memory");
    } else {
      asm volatile("s_wait_asynccnt 0x0" ::: "memory");
    }
    __syncthreads();
    const _Float16* ab = atile[i & 1];
#pragma unroll
    for (int kc = 0; kc < 256; kc += 32) {
      v16h a = frag_ld_kcontig(ab + kc, 256);
#pragma unroll
      for (int j = 0; j < 4; ++j) {
        v16h bb = frag_ld_kcontig(apb + (size_t)(v0 + 16 * j) * LL + pc + kc, LL);
        acc[j] = wmma_f16(a, bb, acc[j]);
      }
    }
    __syncthreads();
  }
  int lane = t & 31, n = lane & 15, mb = (lane & 16) ? 8 : 0;
  const float* iv = invb + (size_t)b * QS + qt * 16;
  float sc[8];
#pragma unroll
  for (int r = 0; r < 8; ++r) sc[r] = iv[mb + r];
  float* dp = pv + ((size_t)b * LL + q0 + (size_t)qt * 16) * VD + v0 + n;
#pragma unroll
  for (int j = 0; j < 4; ++j)
#pragma unroll
    for (int r = 0; r < 8; ++r)
      dp[(size_t)(mb + r) * VD + j * 16] = acc[j][r] * sc[r];
}

// ---------------- K7: overlap-add gather (each pixel sums <=4 taps) ----------
__global__ void k_gather(const float* __restrict__ pv, float* __restrict__ y) {
  const long total = (long)NB * CO * HH * WW;
  for (long i = (long)blockIdx.x * blockDim.x + threadIdx.x; i < total;
       i += (long)gridDim.x * blockDim.x) {
    int X = (int)(i & 127), Y = (int)((i >> 7) & 127);
    int c = (int)((i >> 14) & 63), b = (int)(i >> 20);
    float s = 0.0f;
#pragma unroll
    for (int ky = 0; ky < 4; ++ky) {
      int ty = Y + 1 - ky;
      if (ty < 0 || (ty & 1)) continue;
      int ii = ty >> 1;
      if (ii >= HS) continue;
#pragma unroll
      for (int kx = 0; kx < 4; ++kx) {
        int tx = X + 1 - kx;
        if (tx < 0 || (tx & 1)) continue;
        int jj = tx >> 1;
        if (jj >= WS) continue;
        int q = ii * WS + jj;
        s += pv[((size_t)b * LL + q) * VD + c * 16 + ky * 4 + kx];
      }
    }
    y[i] = s;
  }
}

// ---------------- K8: y2 = Wc @ y (WMMA 64x64) + BN channel stats ------------
__global__ void k_wc(const float* __restrict__ y, const _Float16* __restrict__ wch,
                     float* __restrict__ y2, float* __restrict__ bns) {
  __shared__ _Float16 yt[16 * CO];           // [pix][c]
  int b = blockIdx.y;
  int p0 = blockIdx.x * 16;
  int t = threadIdx.x;                       // 128 threads, 4 waves
  const float* yb = y + (size_t)b * CO * (HH * WW);
  for (int i = t; i < 16 * CO; i += 128) {
    int pix = i >> 6, c = i & 63;
    yt[i] = (_Float16)yb[(size_t)c * (HH * WW) + p0 + pix];
  }
  __syncthreads();
  int wv = t >> 5;
  int o0 = wv * 16;
  v8f acc = {};
#pragma unroll
  for (int kc = 0; kc < CO; kc += 32) {
    v16h a  = frag_ld_kcontig(wch + (size_t)o0 * CO + kc, CO);
    v16h bb = frag_ld_kcontig(yt + kc, CO);
    acc = wmma_f16(a, bb, acc);
  }
  int lane = t & 31, n = lane & 15, mb = (lane & 16) ? 8 : 0;
  float* dp = y2 + ((size_t)(b * CO) + o0 + mb) * (HH * WW) + p0 + n;
  float psum[8], psq[8];
#pragma unroll
  for (int r = 0; r < 8; ++r) {
    float v = acc[r];
    dp[(size_t)r * (HH * WW)] = v;
    psum[r] = v;
    psq[r] = v * v;
  }
#pragma unroll
  for (int m = 1; m < 16; m <<= 1)
#pragma unroll
    for (int r = 0; r < 8; ++r) {
      psum[r] += __shfl_xor(psum[r], m, 32);
      psq[r]  += __shfl_xor(psq[r], m, 32);
    }
  if (n == 0)
#pragma unroll
    for (int r = 0; r < 8; ++r) {
      int o = o0 + mb + r;
      atomicAdd(&bns[o], psum[r]);
      atomicAdd(&bns[CO + o], psq[r]);
    }
}

// ---------------- K9: batchnorm finalize + residual --------------------------
__global__ void k_fin(const float* __restrict__ y2, const float* __restrict__ bns,
                      const float* __restrict__ gamma, const float* __restrict__ beta,
                      const float* __restrict__ alpha, float* __restrict__ out) {
  const long total = (long)NB * CO * HH * WW;
  const float Ninv = 1.0f / (float)(NB * HH * WW);
  for (long i = (long)blockIdx.x * blockDim.x + threadIdx.x; i < total;
       i += (long)gridDim.x * blockDim.x) {
    int c = (int)((i >> 14) & 63);
    float mu  = bns[c] * Ninv;
    float var = bns[CO + c] * Ninv - mu * mu;
    out[i] = gamma[c] * (y2[i] - mu) * rsqrtf(var + 1e-5f) + beta[c] + alpha[i];
  }
}

// ---------------- launch ------------------------------------------------------
extern "C" void kernel_launch(void* const* d_in, const int* in_sizes, int n_in,
                              void* d_out, int out_size, void* d_ws, size_t ws_size,
                              hipStream_t stream) {
  (void)in_sizes; (void)n_in; (void)out_size; (void)ws_size;
  const float* f     = (const float*)d_in[0];
  const float* alpha = (const float*)d_in[1];
  const float* gw    = (const float*)d_in[2];
  const float* gb    = (const float*)d_in[3];
  const float* Wc    = (const float*)d_in[4];
  const float* gma   = (const float*)d_in[5];
  const float* bta   = (const float*)d_in[6];
  float* out = (float*)d_out;

  char* ws = (char*)d_ws;
  size_t off = 0;
  float*    fd   = (float*)(ws + off);    off += (size_t)NB * CO * LL * 4;       // 4 MiB
  _Float16* wp   = (_Float16*)(ws + off); off += (size_t)NB * LL * K9 * 2;       // 18.9 MB
  _Float16* wn   = (_Float16*)(ws + off); off += (size_t)NB * LL * K9 * 2;       // 18.9 MB
  _Float16* apT  = (_Float16*)(ws + off); off += (size_t)NB * VD * LL * 2;       // 33.5 MB
  float*    sbuf = (float*)(ws + off);    off += (size_t)NB * LL * QS * 4;       // 33.5 MB
  _Float16* at   = (_Float16*)(ws + off); off += (size_t)NB * QS * LL * 2;       // 16.8 MB
  float*    pv   = (float*)(ws + off);    off += (size_t)NB * LL * VD * 4;       // 67 MB
  float*    yb   = (float*)(ws + off);    off += (size_t)NB * CO * HH * WW * 4;  // 16.8 MB
  float*    y2   = (float*)(ws + off);    off += (size_t)NB * CO * HH * WW * 4;  // 16.8 MB
  _Float16* gwh  = (_Float16*)(ws + off); off += (size_t)CO * CG * 2;
  _Float16* wch  = (_Float16*)(ws + off); off += (size_t)CO * CO * 2;
  float*    invb = (float*)(ws + off);    off += (size_t)NB * QS * 4;
  float*    bns  = (float*)(ws + off);    off += 2 * CO * 4;

  k_init<<<32, 256, 0, stream>>>(gw, Wc, gwh, wch, bns);
  k_fd<<<dim3(LL / 16, NB), 128, 0, stream>>>(f, gwh, gb, fd);
  k_patch<<<dim3(LL, NB), 256, 0, stream>>>(fd, wp, wn);
  k_ap<<<4096, 256, 0, stream>>>(alpha, apT);

  for (int s = 0; s < NSTR; ++s) {
    int q0 = s * QS;
    k_qk<<<dim3(16, QS / 16, NB), 256, 0, stream>>>(wn, wp, sbuf, q0);
    k_sm<<<dim3(QS / 16, NB), 256, 0, stream>>>(sbuf, at, invb);
    k_pv<<<dim3(2, QS / 16, NB), 256, 0, stream>>>(at, apT, invb, pv, q0);
  }

  k_gather<<<4096, 256, 0, stream>>>(pv, yb);
  k_wc<<<dim3((HH * WW) / 16, NB), 128, 0, stream>>>(yb, wch, y2, bns);
  k_fin<<<4096, 256, 0, stream>>>(y2, bns, gma, bta, alpha, out);
}
